// Attention_47261820125787
// MI455X (gfx1250) — matrix operand order repaired and checked by reading, not compiled, not run on hardware
//
#include <hip/hip_runtime.h>
#include <stdint.h>

// ---------------------------------------------------------------------------
// Problem constants (from reference): B=16, S=1024, H=1024, 16 heads x 64 dim
// ---------------------------------------------------------------------------
#define HDIM   1024
#define ROWS   16384   // B*S
#define NHEAD  16
#define HEADD  64
#define SEQLEN 1024
#define NBATCH 16

typedef __attribute__((ext_vector_type(16))) __bf16 v16bf;
typedef __attribute__((ext_vector_type(8)))  __bf16 v8bf;
typedef __attribute__((ext_vector_type(8)))  float  v8f;

static __device__ __forceinline__ unsigned short f2bf_u16(float f) {
  union { float f; unsigned u; } c; c.f = f;
  unsigned u = c.u;
  u += 0x7FFFu + ((u >> 16) & 1u);          // round-to-nearest-even
  return (unsigned short)(u >> 16);
}
static __device__ __forceinline__ __bf16 f2bf(float f) {
  union { unsigned short s; __bf16 b; } c; c.s = f2bf_u16(f); return c.b;
}

union ABFrag { v16bf v; v8bf h[2]; };

#define WMMA_BF16(A, B, C) \
  __builtin_amdgcn_wmma_f32_16x16x32_bf16(false, (A), false, (B), (short)0, (C), false, false)

// Async global->LDS copy, 16B per lane, tracked by ASYNCcnt (no VGPR return).
// GV addressing: LDS byte address (addr[31:0] of the generic shared pointer)
// in the VDST VGPR, 64-bit global address in the VADDR VGPR pair.
static __device__ __forceinline__ void async_ld_b128(unsigned lds_byte, const void* g) {
  asm volatile("global_load_async_to_lds_b128 %0, %1, off"
               :: "v"(lds_byte), "v"((unsigned long long)(size_t)g) : "memory");
}
static __device__ __forceinline__ void wait_async_le4() {
  asm volatile("s_wait_asynccnt 0x4" ::: "memory");
}
static __device__ __forceinline__ void wait_async_0() {
  asm volatile("s_wait_asynccnt 0x0" ::: "memory");
}

// ---------------------------------------------------------------------------
// fp32 -> bf16 conversion of hidden_states (4 elems / thread, 8B packed store)
// ---------------------------------------------------------------------------
__global__ __launch_bounds__(256)
void cvt_f32_bf16_kernel(const float* __restrict__ x, __bf16* __restrict__ y) {
  size_t i = ((size_t)blockIdx.x * 256 + threadIdx.x) * 4;
  float4 v = *(const float4*)(x + i);
  ushort4 o;
  o.x = f2bf_u16(v.x); o.y = f2bf_u16(v.y); o.z = f2bf_u16(v.z); o.w = f2bf_u16(v.w);
  *reinterpret_cast<ushort4*>(y + i) = o;
}

// ---------------------------------------------------------------------------
// Low-rank weight reconstruction: out[n,i] = sum_k LM[n,k]*RM[k,i] + W[n,i]
// K = 64, tiny (0.13 GFLOP per proj) -> plain fp32 FMA loop, bf16 output.
// ---------------------------------------------------------------------------
__global__ __launch_bounds__(256)
void wrec_kernel(const float* __restrict__ W, const float* __restrict__ LM,
                 const float* __restrict__ RM, __bf16* __restrict__ out) {
  int idx = blockIdx.x * 256 + threadIdx.x;   // 1024*1024 total
  int n = idx >> 10, i = idx & 1023;
  float acc = W[idx];
#pragma unroll 8
  for (int k = 0; k < 64; ++k)
    acc += LM[n * 64 + k] * RM[k * 1024 + i];
  out[idx] = f2bf(acc);
}

// ---------------------------------------------------------------------------
// Projection GEMM: y[m,n] = (sum_k X[m,k]*Wt[n,k] + bias[n]) * Fsc[n]
// X row-major [ROWS,1024] bf16, Wt row-major [1024,1024] bf16 (n = out chan).
// Block 256 thr = 8 waves, tile 128(M)x128(N); wave = 64x32 (4x2 wmma tiles).
// Operands read directly from global (everything is L2-resident: x=32MB,
// w=2MB << 192MB L2), in the ISA bf16 A/B lane layouts.
// ---------------------------------------------------------------------------
__global__ __launch_bounds__(256)
void gemm_proj_kernel(const __bf16* __restrict__ X, const __bf16* __restrict__ Wt,
                      const float* __restrict__ bias, const float* __restrict__ Fsc,
                      __bf16* __restrict__ outB, float* __restrict__ outF) {
  const int lane   = threadIdx.x & 31;
  const int wave   = threadIdx.x >> 5;
  const int laneHi = lane >> 4;      // 0/1 lane half-group
  const int lan15  = lane & 15;
  const int rowBase = blockIdx.x * 128 + (wave & 1) * 64;
  const int colBase = blockIdx.y * 128 + (wave >> 1) * 32;

  v8f acc[4][2];
#pragma unroll
  for (int mt = 0; mt < 4; ++mt)
#pragma unroll
    for (int nt = 0; nt < 2; ++nt) acc[mt][nt] = (v8f){};

  for (int k0 = 0; k0 < HDIM; k0 += 32) {
    ABFrag a[4];
#pragma unroll
    for (int mt = 0; mt < 4; ++mt) {
      // A 16x32 bf16: lane half 0 holds K{0..7,16..23}, half 1 K{8..15,24..31}
      const __bf16* p = X + (size_t)(rowBase + mt * 16 + lan15) * HDIM + k0 + laneHi * 8;
      a[mt].h[0] = *(const v8bf*)p;
      a[mt].h[1] = *(const v8bf*)(p + 16);
    }
    ABFrag bw[2];
#pragma unroll
    for (int nt = 0; nt < 2; ++nt) {
      // B 32x16 bf16: lane = column, lane half 0 holds K{0..7,16..23}, half 1 K{8..15,24..31}
      const __bf16* p = Wt + (size_t)(colBase + nt * 16 + lan15) * HDIM + k0 + laneHi * 8;
      bw[nt].h[0] = *(const v8bf*)p;
      bw[nt].h[1] = *(const v8bf*)(p + 16);
    }
#pragma unroll
    for (int mt = 0; mt < 4; ++mt)
#pragma unroll
      for (int nt = 0; nt < 2; ++nt)
        acc[mt][nt] = WMMA_BF16(a[mt].v, bw[nt].v, acc[mt][nt]);
  }

  // Epilogue: C layout -> VGPR j holds row (j + 8*laneHi), col = lane&15
#pragma unroll
  for (int nt = 0; nt < 2; ++nt) {
    int col = colBase + nt * 16 + lan15;
    float bv = bias[col], fv = Fsc[col];
#pragma unroll
    for (int mt = 0; mt < 4; ++mt)
#pragma unroll
      for (int j = 0; j < 8; ++j) {
        int row = rowBase + mt * 16 + j + 8 * laneHi;
        float y = (acc[mt][nt][j] + bv) * fv;
        if (outF) outF[(size_t)row * HDIM + col] = y;
        else      outB[(size_t)row * HDIM + col] = f2bf(y);
      }
  }
}

// ---------------------------------------------------------------------------
// Flash attention: one workgroup (4 waves / 128 thr) = 64 q-rows of one
// (batch, head). Scores live in registers in native WMMA C layout; running
// max/sum via shfl_xor over the 16-lane half-groups.
// V tiles are prefetched with double-buffered global_load_async_to_lds_b128
// (ASYNCcnt) so the next tile's HBM/L2 latency overlaps the current tile's
// compute; each thread's async region is read back only by that thread, so
// only s_wait_asynccnt + the existing Vt barriers are needed. V is then
// transposed LDS->LDS so P*V B-fragments are contiguous ds_load_b128.
// ---------------------------------------------------------------------------
__global__ __launch_bounds__(128)
void flash_attn_kernel(const __bf16* __restrict__ Q, const __bf16* __restrict__ Kt,
                       const __bf16* __restrict__ V, __bf16* __restrict__ O) {
  __shared__ __align__(16) __bf16 Vraw[2][64][64];  // raw V tiles (async dest)
  __shared__ __align__(16) __bf16 Vt[64][80];       // [d][key], padded stride
  __shared__ __align__(16) __bf16 Ps[4][16][64];    // per-wave P tile

  const int lane   = threadIdx.x & 31;
  const int wave   = threadIdx.x >> 5;
  const int laneHi = lane >> 4;
  const int lan15  = lane & 15;
  const int head   = blockIdx.y;
  const size_t rowOff = (size_t)blockIdx.z * SEQLEN;   // batch row base
  const int colOff = head * HEADD;
  const int qbase  = blockIdx.x * 64 + wave * 16;

  // per-thread staging region: 64B of one V row
  const int vkey = threadIdx.x & 63;
  const int vdh  = (threadIdx.x >> 6) * 32;
  const __bf16* Vbase = V + rowOff * HDIM + colOff;   // (key row, d col)

  // Q fragments (16 rows x 64 dim = 2 K-steps), held in registers
  ABFrag qf[2];
#pragma unroll
  for (int ks = 0; ks < 2; ++ks) {
    const __bf16* p = Q + (rowOff + qbase + lan15) * HDIM + colOff + ks * 32 + laneHi * 8;
    qf[ks].h[0] = *(const v8bf*)p;
    qf[ks].h[1] = *(const v8bf*)(p + 16);
  }

  v8f o[4];
#pragma unroll
  for (int dt = 0; dt < 4; ++dt) o[dt] = (v8f){};
  float m[8], l[8];
#pragma unroll
  for (int j = 0; j < 8; ++j) { m[j] = -1e30f; l[j] = 0.f; }

  // issue async staging of tile 0
  {
    const __bf16* g = Vbase + (size_t)vkey * HDIM + vdh;
    unsigned lb = (unsigned)(size_t)&Vraw[0][vkey][vdh];
#pragma unroll
    for (int c = 0; c < 4; ++c) async_ld_b128(lb + c * 16, g + c * 8);
  }

  for (int it = 0; it < SEQLEN / 64; ++it) {
    const int kb  = it * 64;
    const int buf = it & 1;

    // prefetch next tile into the other buffer
    if (it + 1 < SEQLEN / 64) {
      const __bf16* g = Vbase + (size_t)(kb + 64 + vkey) * HDIM + vdh;
      unsigned lb = (unsigned)(size_t)&Vraw[buf ^ 1][vkey][vdh];
#pragma unroll
      for (int c = 0; c < 4; ++c) async_ld_b128(lb + c * 16, g + c * 8);
      wait_async_le4();   // current tile done; next tile (4 ops) in flight
    } else {
      wait_async_0();
    }

    // ---- transpose own region: Vraw[buf] -> Vt ----
    {
      const __bf16* r = &Vraw[buf][vkey][vdh];
      v8bf tmp[4];
#pragma unroll
      for (int c = 0; c < 4; ++c) tmp[c] = *(const v8bf*)(r + c * 8);
#pragma unroll
      for (int c = 0; c < 4; ++c)
#pragma unroll
        for (int e = 0; e < 8; ++e)
          Vt[vdh + c * 8 + e][vkey] = tmp[c][e];
    }
    __syncthreads();

    // ---- scores: 16 q x 64 keys (4 n-tiles, 2 K-steps over head dim) ----
    v8f s[4];
#pragma unroll
    for (int nt = 0; nt < 4; ++nt) {
      s[nt] = (v8f){};
#pragma unroll
      for (int ks = 0; ks < 2; ++ks) {
        ABFrag bk;
        const __bf16* p = Kt + (rowOff + kb + nt * 16 + lan15) * HDIM + colOff + ks * 32 + laneHi * 8;
        bk.h[0] = *(const v8bf*)p;
        bk.h[1] = *(const v8bf*)(p + 16);
        s[nt] = WMMA_BF16(qf[ks].v, bk.v, s[nt]);
      }
    }

    // ---- online softmax (rows live in half-groups of 16 lanes) ----
#pragma unroll
    for (int j = 0; j < 8; ++j) {
      float mx = -1e30f;
#pragma unroll
      for (int nt = 0; nt < 4; ++nt) { s[nt][j] *= 0.125f; mx = fmaxf(mx, s[nt][j]); }
      mx = fmaxf(mx, __shfl_xor(mx, 1, 32));
      mx = fmaxf(mx, __shfl_xor(mx, 2, 32));
      mx = fmaxf(mx, __shfl_xor(mx, 4, 32));
      mx = fmaxf(mx, __shfl_xor(mx, 8, 32));
      float mnew = fmaxf(m[j], mx);
      float corr = __expf(m[j] - mnew);
      float rsum = 0.f;
#pragma unroll
      for (int nt = 0; nt < 4; ++nt) {
        float pv = __expf(s[nt][j] - mnew);
        s[nt][j] = pv;
        rsum += pv;
      }
      rsum += __shfl_xor(rsum, 1, 32);
      rsum += __shfl_xor(rsum, 2, 32);
      rsum += __shfl_xor(rsum, 4, 32);
      rsum += __shfl_xor(rsum, 8, 32);
      l[j] = l[j] * corr + rsum;
      m[j] = mnew;
#pragma unroll
      for (int dt = 0; dt < 4; ++dt) o[dt][j] *= corr;
    }

    // ---- relay P (C layout) -> A layout via per-wave LDS ----
#pragma unroll
    for (int nt = 0; nt < 4; ++nt)
#pragma unroll
      for (int j = 0; j < 8; ++j)
        Ps[wave][j + 8 * laneHi][nt * 16 + lan15] = f2bf(s[nt][j]);

    ABFrag pf[2];
#pragma unroll
    for (int ks = 0; ks < 2; ++ks) {
      const __bf16* p = &Ps[wave][lan15][ks * 32 + laneHi * 8];
      pf[ks].h[0] = *(const v8bf*)p;
      pf[ks].h[1] = *(const v8bf*)(p + 16);
    }

    // ---- ctx += P * V ----
#pragma unroll
    for (int dt = 0; dt < 4; ++dt)
#pragma unroll
      for (int ks = 0; ks < 2; ++ks) {
        ABFrag bv;
        const __bf16* p = &Vt[dt * 16 + lan15][ks * 32 + laneHi * 8];
        bv.h[0] = *(const v8bf*)p;
        bv.h[1] = *(const v8bf*)(p + 16);
        o[dt] = WMMA_BF16(pf[ks].v, bv.v, o[dt]);
      }
    __syncthreads();  // protect Vt before next block overwrites it
  }

  // ---- normalize + store ctx (bf16, row-major [ROWS, HDIM]) ----
#pragma unroll
  for (int dt = 0; dt < 4; ++dt)
#pragma unroll
    for (int j = 0; j < 8; ++j) {
      int row = qbase + j + 8 * laneHi;
      O[(rowOff + row) * HDIM + colOff + dt * 16 + lan15] = f2bf(o[dt][j] / l[j]);
    }
}

// ---------------------------------------------------------------------------
// Launch: cvt -> 4x wrec -> q/k/v GEMMs -> flash attention -> output GEMM
// Workspace: xb 32MB + 4 weights 8MB + q/k/v 96MB + ctx 32MB = 168MB
// ---------------------------------------------------------------------------
extern "C" void kernel_launch(void* const* d_in, const int* in_sizes, int n_in,
                              void* d_out, int out_size, void* d_ws, size_t ws_size,
                              hipStream_t stream) {
  (void)in_sizes; (void)n_in; (void)out_size; (void)ws_size;
  const float* hs = (const float*)d_in[0];
  const float* W[4]  = {(const float*)d_in[1],  (const float*)d_in[6],
                        (const float*)d_in[11], (const float*)d_in[16]};
  const float* bB[4] = {(const float*)d_in[2],  (const float*)d_in[7],
                        (const float*)d_in[12], (const float*)d_in[17]};
  const float* LM[4] = {(const float*)d_in[3],  (const float*)d_in[8],
                        (const float*)d_in[13], (const float*)d_in[18]};
  const float* RM[4] = {(const float*)d_in[4],  (const float*)d_in[9],
                        (const float*)d_in[14], (const float*)d_in[19]};
  const float* Fs[4] = {(const float*)d_in[5],  (const float*)d_in[10],
                        (const float*)d_in[15], (const float*)d_in[20]};

  uint8_t* p = (uint8_t*)d_ws;
  __bf16* xb = (__bf16*)p; p += (size_t)ROWS * HDIM * 2;
  __bf16* wr[4];
  for (int i = 0; i < 4; ++i) { wr[i] = (__bf16*)p; p += (size_t)HDIM * HDIM * 2; }
  __bf16* qb = (__bf16*)p; p += (size_t)ROWS * HDIM * 2;
  __bf16* kb = (__bf16*)p; p += (size_t)ROWS * HDIM * 2;
  __bf16* vb = (__bf16*)p; p += (size_t)ROWS * HDIM * 2;
  __bf16* cb = (__bf16*)p; p += (size_t)ROWS * HDIM * 2;

  cvt_f32_bf16_kernel<<<(ROWS * HDIM) / 1024, 256, 0, stream>>>(hs, xb);
  for (int i = 0; i < 4; ++i)
    wrec_kernel<<<(HDIM * HDIM) / 256, 256, 0, stream>>>(W[i], LM[i], RM[i], wr[i]);

  dim3 gg(ROWS / 128, HDIM / 128);
  gemm_proj_kernel<<<gg, 256, 0, stream>>>(xb, wr[0], bB[0], Fs[0], qb, nullptr);
  gemm_proj_kernel<<<gg, 256, 0, stream>>>(xb, wr[1], bB[1], Fs[1], kb, nullptr);
  gemm_proj_kernel<<<gg, 256, 0, stream>>>(xb, wr[2], bB[2], Fs[2], vb, nullptr);

  dim3 ga(SEQLEN / 64, NHEAD, NBATCH);
  flash_attn_kernel<<<ga, 128, 0, stream>>>(qb, kb, vb, cb);

  gemm_proj_kernel<<<gg, 256, 0, stream>>>(cb, wr[3], bB[3], Fs[3], nullptr, (float*)d_out);
}
